// HeteroRelConv_41764261986620
// MI455X (gfx1250) — matrix operand order repaired and checked by reading, not compiled, not run on hardware
//
#include <hip/hip_runtime.h>
#include <hip/hip_bf16.h>

typedef __attribute__((ext_vector_type(16))) __bf16 v16bf;
typedef __attribute__((ext_vector_type(8)))  float  v8f;

// ---------------------------------------------------------------- helpers
__device__ __forceinline__ unsigned short f2bf_bits(float x) {
    unsigned u = __builtin_bit_cast(unsigned, x);
    unsigned r = (u + 0x7FFFu + ((u >> 16) & 1u)) >> 16;   // round-to-nearest-even
    return (unsigned short)r;
}
__device__ __forceinline__ __bf16 f2bf(float x) {
    unsigned short b = f2bf_bits(x);
    return __builtin_bit_cast(__bf16, b);
}
__device__ __forceinline__ void atomic_add_f32(float* p, float v) {
    __hip_atomic_fetch_add(p, v, __ATOMIC_RELAXED, __HIP_MEMORY_SCOPE_AGENT);
}

// ---------------------------------------------------------------- B packing
// Pack Wf/Ws (rows r=6,7,8 of [9,128,64] fp32) into WMMA B-operand layout:
// per (rel, mat, kc, nt) tile of 32x16 bf16:
//   lane L: n = nt*16 + L%16 ; K group = (L/16)*16
//   dword v (0..7): K = kc*32 + group + 2v (lo half), +1 (hi half)
// Storage: Bpack[rel][mat][kc*4+nt][lane][v]  (dwords)
__global__ __launch_bounds__(32)
void pack_B_kernel(const float* __restrict__ Wf, const float* __restrict__ Ws,
                   unsigned* __restrict__ Bpack) {
    int lane = threadIdx.x;                 // 0..31
    int blk  = blockIdx.x;                  // 0..95
    int rel  = blk >> 5;                    // 0..2  -> relation r = 6+rel
    int rem  = blk & 31;
    int mat  = rem >> 4;                    // 0 = Wf, 1 = Ws
    int t16  = rem & 15;                    // kc*4 + nt
    int kc   = t16 >> 2, nt = t16 & 3;
    const float* W = (mat == 0 ? Wf : Ws) + (size_t)(6 + rel) * 128 * 64;
    int n  = nt * 16 + (lane & 15);
    int kg = (lane >> 4) * 16;
    unsigned* out = Bpack + ((((size_t)rel * 2 + mat) * 16 + t16) * 32 + lane) * 8;
#pragma unroll
    for (int v = 0; v < 8; ++v) {
        int k = kc * 32 + kg + 2 * v;
        unsigned lo = f2bf_bits(W[(size_t)k * 64 + n]);
        unsigned hi = f2bf_bits(W[(size_t)(k + 1) * 64 + n]);
        out[v] = lo | (hi << 16);
    }
}

// ---------------------------------------------------------------- input projections
// out[row] = bf16( X[row] @ W + b ),  X: [N,K] fp32, W: [K,64], 4 rows per block.
__global__ __launch_bounds__(256)
void proj_kernel(const float* __restrict__ X, int K,
                 const float* __restrict__ W, const float* __restrict__ b,
                 __bf16* __restrict__ out) {
    __shared__ float sh[4][92];
    int g = threadIdx.x >> 6;               // row group 0..3
    int c = threadIdx.x & 63;               // output column
    long row = (long)blockIdx.x * 4 + g;
    for (int i = c; i < K; i += 64) sh[g][i] = X[row * K + i];
    __syncthreads();
    float acc = b[c];
    for (int k = 0; k < K; ++k) acc += sh[g][k] * W[k * 64 + c];
    out[row * 64 + c] = f2bf(acc);
}

// cell: f32 accumulator init (3 incoming relations => 3*x) + bf16 copy
__global__ __launch_bounds__(256)
void cell_init_kernel(const float* __restrict__ xc, float* __restrict__ outc,
                      __bf16* __restrict__ xcb) {
    int i = blockIdx.x * 256 + threadIdx.x;
    float v = xc[i];
    outc[i] = 3.0f * v;
    xcb[i]  = f2bf(v);
}

// ---------------------------------------------------------------- CGConv edge kernel
// One wave32 per 16-edge tile. z = [x_dst | x_src] in R^128 (bf16),
// gate = sigmoid(z@Wf+bf), val = softplus(z@Ws+bs), scatter-add gate*val to out[dst].
__global__ __launch_bounds__(256)
void edge_cgconv_wmma(const int* __restrict__ e, int E,
                      const __bf16* __restrict__ xsrc,  // [Ns,64] bf16
                      const __bf16* __restrict__ xdst,  // [Nd,64] bf16
                      float* __restrict__ outdst,       // [Nd,64] f32 accum
                      const uint4* __restrict__ Bf,     // packed 16 tiles * 32 lanes * 2
                      const uint4* __restrict__ Bs,
                      const float* __restrict__ biasf,  // [64]
                      const float* __restrict__ biass) {
    int lane  = threadIdx.x & 31;
    int wave  = threadIdx.x >> 5;
    int tile  = blockIdx.x * 8 + wave;
    int ebase = tile * 16;
    int m     = lane & 15;
    int srcn  = e[ebase + m];
    int dstn  = e[E + ebase + m];
    int hi8   = (lane >> 4) << 4;           // 0 or 16 (A-operand K sub-group)

    v8f accF[4] = {};
    v8f accS[4] = {};
    const __bf16* rowi = xdst + (long)dstn * 64;   // x_i (target) -> z[0:64]
    const __bf16* rowj = xsrc + (long)srcn * 64;   // x_j (source) -> z[64:128]

#pragma unroll
    for (int kc = 0; kc < 4; ++kc) {
        const __bf16* row = (kc < 2) ? rowi : rowj;
        int colbase = ((kc & 1) << 5) + hi8;
        union { v16bf v; uint4 q[2]; } a;
        a.q[0] = *(const uint4*)(row + colbase);        // K = base .. base+7
        a.q[1] = *(const uint4*)(row + colbase + 8);    // K = base+8 .. base+15
#pragma unroll
        for (int nt = 0; nt < 4; ++nt) {
            union { v16bf v; uint4 q[2]; } bF, bS;
            long bidx = (((long)(kc * 4 + nt)) * 32 + lane) * 2;
            bF.q[0] = Bf[bidx]; bF.q[1] = Bf[bidx + 1];
            bS.q[0] = Bs[bidx]; bS.q[1] = Bs[bidx + 1];
            accF[nt] = __builtin_amdgcn_wmma_f32_16x16x32_bf16(
                false, a.v, false, bF.v, (short)0, accF[nt], false, false);
            accS[nt] = __builtin_amdgcn_wmma_f32_16x16x32_bf16(
                false, a.v, false, bS.v, (short)0, accS[nt], false, false);
        }
    }

    // Epilogue: C/D layout: lane -> (n = lane%16), dword v -> row m = v + 8*(lane/16)
    int n   = lane & 15;
    int mhi = (lane >> 4) << 3;
#pragma unroll
    for (int nt = 0; nt < 4; ++nt) {
        int col = nt * 16 + n;
        float bfv = biasf[col], bsv = biass[col];
#pragma unroll
        for (int v = 0; v < 8; ++v) {
            int mm = mhi + v;
            int dn = __shfl(dstn, mm, 32);      // dst node of tile-row mm
            float f = accF[nt][v] + bfv;
            float s = accS[nt][v] + bsv;
            float g  = 1.f / (1.f + __expf(-f));
            float sp = (s > 20.f) ? s : __logf(1.f + __expf(s));
            atomic_add_f32(outdst + (long)dn * 64 + col, g * sp);
        }
    }
}

// ---------------------------------------------------------------- head
// y[row] = softplus(relu(outc[row]) @ Wp + bp) @ Wo + bo
__global__ __launch_bounds__(64)
void head_kernel(const float* __restrict__ outc, const float* __restrict__ Wp,
                 const float* __restrict__ bp, const float* __restrict__ Wo,
                 const float* __restrict__ bo, float* __restrict__ y) {
    __shared__ float xr[64];
    __shared__ float red[64];
    int c = threadIdx.x;
    long row = blockIdx.x;
    float v = outc[row * 64 + c];
    xr[c] = v > 0.f ? v : 0.f;
    __syncthreads();
    float acc = bp[c];
    for (int k = 0; k < 64; ++k) acc += xr[k] * Wp[k * 64 + c];
    float sp = (acc > 20.f) ? acc : __logf(1.f + __expf(acc));
    red[c] = sp * Wo[c];
    __syncthreads();
    for (int s = 32; s > 0; s >>= 1) {
        if (c < s) red[c] += red[c + s];
        __syncthreads();
    }
    if (c == 0) y[row] = red[0] + bo[0];
}

// ---------------------------------------------------------------- launch
extern "C" void kernel_launch(void* const* d_in, const int* in_sizes, int n_in,
                              void* d_out, int out_size, void* d_ws, size_t ws_size,
                              hipStream_t stream) {
    const float* x_atom  = (const float*)d_in[0];
    const float* x_bond  = (const float*)d_in[1];
    const float* x_motif = (const float*)d_in[2];
    const float* x_cell  = (const float*)d_in[3];
    const float* Wa = (const float*)d_in[4];
    const float* ba = (const float*)d_in[5];
    const float* Wb = (const float*)d_in[6];
    const float* bb = (const float*)d_in[7];
    const float* Wm = (const float*)d_in[8];
    const float* bm = (const float*)d_in[9];
    const float* Wf = (const float*)d_in[10];
    const float* bf = (const float*)d_in[11];
    const float* Ws = (const float*)d_in[12];
    const float* bs = (const float*)d_in[13];
    const float* Wp = (const float*)d_in[14];
    const float* bp = (const float*)d_in[15];
    const float* Wo = (const float*)d_in[16];
    const float* bo = (const float*)d_in[17];
    const int* e_ac = (const int*)d_in[24];   // src=atom , dst=cell, E=65536  (r=6)
    const int* e_bc = (const int*)d_in[25];   // src=bond , dst=cell, E=131072 (r=7)
    const int* e_mc = (const int*)d_in[26];   // src=motif, dst=cell, E=16384  (r=8)

    const int NA = 65536, NB = 131072, NM = 16384, NC = 2048;

    // workspace layout (all 256B-aligned)
    char* ws = (char*)d_ws;
    unsigned* Bpack = (unsigned*)(ws + 0);            //  98304 B (3*2*16*32*8 dwords)
    __bf16* xa  = (__bf16*)(ws + 98304);              //  8 MB
    __bf16* xb  = (__bf16*)(ws + 98304 + 8388608);    // 16 MB
    __bf16* xm  = (__bf16*)(ws + 98304 + 25165824);   //  2 MB
    __bf16* xcb = (__bf16*)(ws + 98304 + 27262976);   // 256 KB
    float*  outc = (float*)(ws + 98304 + 27525120);   // 512 KB

    // 1) weight packing + projections + cell init (independent)
    pack_B_kernel<<<96, 32, 0, stream>>>(Wf, Ws, Bpack);
    proj_kernel<<<NA / 4, 256, 0, stream>>>(x_atom,  92, Wa, ba, xa);
    proj_kernel<<<NB / 4, 256, 0, stream>>>(x_bond,   7, Wb, bb, xb);
    proj_kernel<<<NM / 4, 256, 0, stream>>>(x_motif, 35, Wm, bm, xm);
    cell_init_kernel<<<(NC * 64) / 256, 256, 0, stream>>>(x_cell, outc, xcb);

    // 2) the three live relations (dst == cell); per-relation packed-B offsets
    const uint4* Bq = (const uint4*)Bpack;            // 1024 uint4 per (rel,mat)
    edge_cgconv_wmma<<<65536 / 128, 256, 0, stream>>>(
        e_ac, 65536,  xa, xcb, outc, Bq + 0 * 2048,        Bq + 0 * 2048 + 1024,
        bf + 6 * 64, bs + 6 * 64);
    edge_cgconv_wmma<<<131072 / 128, 256, 0, stream>>>(
        e_bc, 131072, xb, xcb, outc, Bq + 1 * 2048,        Bq + 1 * 2048 + 1024,
        bf + 7 * 64, bs + 7 * 64);
    edge_cgconv_wmma<<<16384 / 128, 256, 0, stream>>>(
        e_mc, 16384,  xm, xcb, outc, Bq + 2 * 2048,        Bq + 2 * 2048 + 1024,
        bf + 8 * 64, bs + 8 * 64);

    // 3) head -> d_out [2048]
    head_kernel<<<NC, 64, 0, stream>>>(outc, Wp, bp, Wo, bo, (float*)d_out);
}
